// PolyAttention_61289183314195
// MI455X (gfx1250) — matrix operand order repaired and checked by reading, not compiled, not run on hardware
//
#include <hip/hip_runtime.h>
#include <hip/hip_bf16.h>

// ---- problem constants --------------------------------------------------
constexpr int BS   = 4;
constexpr int NSEQ = 2048;
constexpr int DIMC = 768;
constexpr int NH   = 12;
constexpr int HD   = 64;          // head dim
constexpr int BN   = BS * NSEQ;   // 8192 rows
constexpr int OC3  = 3 * DIMC;    // 2304
#define SCALE_F 0.125f            // HD^-0.5

// ---- WMMA vector types --------------------------------------------------
typedef _Float16 v16h __attribute__((ext_vector_type(16)));
typedef _Float16 v8h  __attribute__((ext_vector_type(8)));
typedef float    v8f  __attribute__((ext_vector_type(8)));

static __device__ __forceinline__ v16h cat8(v8h lo, v8h hi) {
    v16h r;
#pragma unroll
    for (int i = 0; i < 8; ++i) { r[i] = lo[i]; r[8 + i] = hi[i]; }
    return r;
}

static __device__ __forceinline__ v8f wmma_f16(v16h a, v16h b, v8f c) {
    return __builtin_amdgcn_wmma_f32_16x16x32_f16(false, a, false, b,
                                                  (short)0, c, false, false);
}

// A-fragment (16x32 f16): lane M = lane&15, half = lane>>4.
// halves [0..7] = K k0+half*8+0..7, [8..15] = K k0+16+half*8+0..7.
static __device__ __forceinline__ v16h load_a_frag(const _Float16* __restrict__ p,
                                                   int row0, long ld, int k0,
                                                   int lane) {
    const int M = lane & 15, half = lane >> 4;
    const _Float16* q = p + (long)(row0 + M) * ld + k0 + half * 8;
    v8h lo = *(const v8h*)q;
    v8h hi = *(const v8h*)(q + 16);
    return cat8(lo, hi);
}

// B-fragment (32x16 f16) from a K-major buffer:
// element B[k][n] at  p + (col0+n)*ld + k0 + k; halves [0..7] = K half*8+0..7, [8..15] = K 16+half*8+0..7.
static __device__ __forceinline__ v16h load_b_frag(const _Float16* __restrict__ p,
                                                   int col0, long ld, int k0,
                                                   int lane) {
    const int n = lane & 15, half = lane >> 4;
    const _Float16* q = p + (long)(col0 + n) * ld + k0 + half * 8;
    return cat8(*(const v8h*)q, *(const v8h*)(q + 16));
}

// ---- kernel 0: fp32 -> f16 convert -------------------------------------
__global__ void cvt_f32_f16(const float* __restrict__ in,
                            _Float16* __restrict__ out, int n) {
    int i = blockIdx.x * blockDim.x + threadIdx.x;
    if (i < n) out[i] = (_Float16)in[i];
}

// ---- kernel 1: QKV GEMM  (x @ w_qkv^T), scatter into q / k / v^T -------
// grid: (BN/128, OC3/64), block 128 (4 waves); wave tile = 32(M) x 64(N)
__global__ __launch_bounds__(128) void qkv_gemm(
        const _Float16* __restrict__ xh, const _Float16* __restrict__ wh,
        _Float16* __restrict__ qh, _Float16* __restrict__ kh,
        _Float16* __restrict__ vth) {
    const int lane = threadIdx.x & 31;
    const int wave = threadIdx.x >> 5;
    const int m0  = blockIdx.x * 128 + wave * 32;  // row (b*n) tile
    const int oc0 = blockIdx.y * 64;               // output-channel tile
    const int nn = lane & 15, half = lane >> 4;

    v8f acc[2][4] = {{{}, {}, {}, {}}, {{}, {}, {}, {}}};
    for (int c0 = 0; c0 < DIMC; c0 += 32) {
        v16h a0 = load_a_frag(xh, m0,      DIMC, c0, lane);
        v16h a1 = load_a_frag(xh, m0 + 16, DIMC, c0, lane);
#pragma unroll
        for (int t = 0; t < 4; ++t) {
            v16h b = load_b_frag(wh, oc0 + t * 16, DIMC, c0, lane);
            acc[0][t] = wmma_f16(a0, b, acc[0][t]);
            acc[1][t] = wmma_f16(a1, b, acc[1][t]);
        }
    }
    // scatter C tiles (rows = half*8+r, col = nn) into per-head buffers
#pragma unroll
    for (int u = 0; u < 2; ++u) {
#pragma unroll
        for (int t = 0; t < 4; ++t) {
            const int oc = oc0 + t * 16 + nn;
#pragma unroll
            for (int r = 0; r < 8; ++r) {
                const int bn = m0 + u * 16 + half * 8 + r;
                const int bb = bn >> 11;           // /2048
                const int n  = bn & (NSEQ - 1);
                const _Float16 hv = (_Float16)acc[u][t][r];
                if (oc < DIMC) {                       // Q: [b,h,n,d]
                    const int h = oc >> 6, d = oc & 63;
                    qh[(((long)bb * NH + h) * NSEQ + n) * HD + d] = hv;
                } else if (oc < 2 * DIMC) {            // K: [b,h,n,d]
                    const int o2 = oc - DIMC, h = o2 >> 6, d = o2 & 63;
                    kh[(((long)bb * NH + h) * NSEQ + n) * HD + d] = hv;
                } else {                               // V^T: [b,h,d,n]
                    const int o2 = oc - 2 * DIMC, h = o2 >> 6, d = o2 & 63;
                    vth[(((long)bb * NH + h) * HD + d) * NSEQ + n] = hv;
                }
            }
        }
    }
}

// ---- kernel 2: fused squared-score attention ---------------------------
// grid: (NSEQ/128, BS*NH), block 128; each wave owns a 32-query tile
// (two 16-row WMMA tiles) and streams all 2048 keys in chunks of 32,
// reusing each K/V fragment for both query tiles (16 WMMA / 8 loads).
__global__ __launch_bounds__(128) void attn_fused(
        const _Float16* __restrict__ qh, const _Float16* __restrict__ kh,
        const _Float16* __restrict__ vth, _Float16* __restrict__ ah) {
    __shared__ _Float16 smem[4 * 2 * 16 * 32];    // 2KB slice per wave
    const int lane = threadIdx.x & 31;
    const int wave = threadIdx.x >> 5;
    const int nn = lane & 15, half = lane >> 4;
    const int bh = blockIdx.y;                    // 0..47
    const int b = bh / NH, h = bh % NH;
    const int q0 = blockIdx.x * 128 + wave * 32;

    const _Float16* qb  = qh  + (long)bh * NSEQ * HD;
    const _Float16* kb  = kh  + (long)bh * NSEQ * HD;
    const _Float16* vtb = vth + (long)bh * HD * NSEQ;
    _Float16* ls = &smem[wave * 2 * 16 * 32];

    // Q A-fragments for both query tiles, K = 0..31 / 32..63 (resident)
    v16h aq[2][2];
#pragma unroll
    for (int qt = 0; qt < 2; ++qt) {
        aq[qt][0] = load_a_frag(qb, q0 + qt * 16, HD, 0,  lane);
        aq[qt][1] = load_a_frag(qb, q0 + qt * 16, HD, 32, lane);
    }

    v8f accd[2][4] = {{{}, {}, {}, {}}, {{}, {}, {}, {}}};
    v8f rs[2] = {{}, {}};             // per-lane partial row sums of s^2

    for (int j0 = 0; j0 < NSEQ; j0 += 32) {
        // K fragments for two 16-key subtiles, shared by both query tiles
        v16h bk00 = load_b_frag(kb, j0,      HD, 0,  lane);
        v16h bk01 = load_b_frag(kb, j0,      HD, 32, lane);
        v16h bk10 = load_b_frag(kb, j0 + 16, HD, 0,  lane);
        v16h bk11 = load_b_frag(kb, j0 + 16, HD, 32, lane);

        v8f sc[2][2] = {{{}, {}}, {{}, {}}};
#pragma unroll
        for (int qt = 0; qt < 2; ++qt) {
            sc[qt][0] = wmma_f16(aq[qt][0], bk00, sc[qt][0]);
            sc[qt][0] = wmma_f16(aq[qt][1], bk01, sc[qt][0]);
            sc[qt][1] = wmma_f16(aq[qt][0], bk10, sc[qt][1]);
            sc[qt][1] = wmma_f16(aq[qt][1], bk11, sc[qt][1]);
        }

        // scale, square, accumulate row sums, stage s^2 tiles in LDS
#pragma unroll
        for (int qt = 0; qt < 2; ++qt) {
#pragma unroll
            for (int r = 0; r < 8; ++r) {
                float s0 = sc[qt][0][r] * SCALE_F; s0 *= s0;
                float s1 = sc[qt][1][r] * SCALE_F; s1 *= s1;
                rs[qt][r] += s0 + s1;
                const int row = half * 8 + r;
                ls[qt * 512 + row * 32 + nn]      = (_Float16)s0;
                ls[qt * 512 + row * 32 + 16 + nn] = (_Float16)s1;
            }
        }
        __syncthreads();
        // re-read as A-fragments (keys become the K dimension)
        v16h as2[2];
#pragma unroll
        for (int qt = 0; qt < 2; ++qt) {
            v8h lo = *(const v8h*)(ls + qt * 512 + nn * 32 + half * 8);
            v8h hi = *(const v8h*)(ls + qt * 512 + nn * 32 + 16 + half * 8);
            as2[qt] = cat8(lo, hi);
        }
        __syncthreads();                           // WAR guard for next iter

        // s^2 @ V : V^T is [d, n] so B-fragments are contiguous loads,
        // each reused by both query tiles
#pragma unroll
        for (int t = 0; t < 4; ++t) {
            v16h bv = load_b_frag(vtb, t * 16, NSEQ, j0, lane);
            accd[0][t] = wmma_f16(as2[0], bv, accd[0][t]);
            accd[1][t] = wmma_f16(as2[1], bv, accd[1][t]);
        }
    }

    // reduce row sums across the 16 columns (lanes within each half)
#pragma unroll
    for (int qt = 0; qt < 2; ++qt) {
#pragma unroll
        for (int r = 0; r < 8; ++r) {
            float v = rs[qt][r];
            v += __shfl_xor(v, 1);
            v += __shfl_xor(v, 2);
            v += __shfl_xor(v, 4);
            v += __shfl_xor(v, 8);
            rs[qt][r] = v;
        }
    }

    // normalize and write to [B, N, C] (channel = h*64 + d) as f16
#pragma unroll
    for (int qt = 0; qt < 2; ++qt) {
#pragma unroll
        for (int t = 0; t < 4; ++t) {
#pragma unroll
            for (int r = 0; r < 8; ++r) {
                const int n = q0 + qt * 16 + half * 8 + r;
                const float o = accd[qt][t][r] / (rs[qt][r] + 1e-6f);
                ah[((long)(b * NSEQ + n)) * DIMC + h * HD + t * 16 + nn] =
                    (_Float16)o;
            }
        }
    }
}

// ---- kernel 3: output projection + bias --------------------------------
// grid: (BN/128, DIMC/64), block 128; wave tile = 32(M) x 64(N), f32 out
__global__ __launch_bounds__(128) void proj_gemm(
        const _Float16* __restrict__ ah, const _Float16* __restrict__ wph,
        const float* __restrict__ bias, float* __restrict__ out) {
    const int lane = threadIdx.x & 31;
    const int wave = threadIdx.x >> 5;
    const int m0  = blockIdx.x * 128 + wave * 32;
    const int oc0 = blockIdx.y * 64;
    const int nn = lane & 15, half = lane >> 4;

    v8f acc[2][4] = {{{}, {}, {}, {}}, {{}, {}, {}, {}}};
    for (int c0 = 0; c0 < DIMC; c0 += 32) {
        v16h a0 = load_a_frag(ah, m0,      DIMC, c0, lane);
        v16h a1 = load_a_frag(ah, m0 + 16, DIMC, c0, lane);
#pragma unroll
        for (int t = 0; t < 4; ++t) {
            v16h b = load_b_frag(wph, oc0 + t * 16, DIMC, c0, lane);
            acc[0][t] = wmma_f16(a0, b, acc[0][t]);
            acc[1][t] = wmma_f16(a1, b, acc[1][t]);
        }
    }
#pragma unroll
    for (int u = 0; u < 2; ++u) {
#pragma unroll
        for (int t = 0; t < 4; ++t) {
            const int oc = oc0 + t * 16 + nn;
            const float bval = bias[oc];
#pragma unroll
            for (int r = 0; r < 8; ++r) {
                const int bn = m0 + u * 16 + half * 8 + r;
                out[(long)bn * DIMC + oc] = acc[u][t][r] + bval;
            }
        }
    }
}

// ---- workspace layout (halves) -----------------------------------------
constexpr long NX  = (long)BN * DIMC;     // 6,291,456
constexpr long NWQ = (long)OC3 * DIMC;    // 1,769,472
constexpr long NWP = (long)DIMC * DIMC;   //   589,824
constexpr long OFF_XH    = 0;
constexpr long OFF_WQKV  = OFF_XH + NX;
constexpr long OFF_WPROJ = OFF_WQKV + NWQ;
constexpr long OFF_Q     = OFF_WPROJ + NWP;
constexpr long OFF_K     = OFF_Q + NX;
constexpr long OFF_VT    = OFF_K + NX;
constexpr long OFF_AH    = OFF_VT + NX;   // total ~67.6 MB of f16

extern "C" void kernel_launch(void* const* d_in, const int* in_sizes, int n_in,
                              void* d_out, int out_size, void* d_ws, size_t ws_size,
                              hipStream_t stream) {
    const float* x      = (const float*)d_in[0];
    const float* w_qkv  = (const float*)d_in[1];
    const float* w_proj = (const float*)d_in[2];
    const float* b_proj = (const float*)d_in[3];
    float* out = (float*)d_out;

    _Float16* ws = (_Float16*)d_ws;
    _Float16* xh    = ws + OFF_XH;
    _Float16* wqkvh = ws + OFF_WQKV;
    _Float16* wprojh= ws + OFF_WPROJ;
    _Float16* qh    = ws + OFF_Q;
    _Float16* kh    = ws + OFF_K;
    _Float16* vth   = ws + OFF_VT;
    _Float16* ah    = ws + OFF_AH;

    cvt_f32_f16<<<(int)((NX  + 255) / 256), 256, 0, stream>>>(x,      xh,    (int)NX);
    cvt_f32_f16<<<(int)((NWQ + 255) / 256), 256, 0, stream>>>(w_qkv,  wqkvh, (int)NWQ);
    cvt_f32_f16<<<(int)((NWP + 255) / 256), 256, 0, stream>>>(w_proj, wprojh,(int)NWP);

    qkv_gemm  <<<dim3(BN / 128, OC3 / 64),  128, 0, stream>>>(xh, wqkvh, qh, kh, vth);
    attn_fused<<<dim3(NSEQ / 128, BS * NH), 128, 0, stream>>>(qh, kh, vth, ah);
    proj_gemm <<<dim3(BN / 128, DIMC / 64), 128, 0, stream>>>(ah, wprojh, b_proj, out);
}
